// ContrastiveGNN_24232205484079
// MI455X (gfx1250) — matrix operand order repaired and checked by reading, not compiled, not run on hardware
//
#include <hip/hip_runtime.h>
#include <hip/hip_bf16.h>

// ---------------- problem constants (match reference) ----------------
#define NN   50000      // nodes
#define NE   800000     // edges
#define NG_G 256        // graphs
#define FIN  128
#define HID  256
#define EMB  128
#define NGRP 16
#define NFAM 128

typedef __attribute__((ext_vector_type(16))) __bf16 v16bf;
typedef __attribute__((ext_vector_type(8)))  __bf16 v8bf;
typedef __attribute__((ext_vector_type(8)))  float  v8f;

static __device__ __forceinline__ __bf16 f2bf(float f) {
    union { float f; unsigned u; } v; v.f = f;
    unsigned r = (v.u + 0x7FFFu + ((v.u >> 16) & 1u)) >> 16;   // round-to-nearest-even
    union { unsigned short s; __bf16 b; } o; o.s = (unsigned short)r;
    return o.b;
}

// ---------------- tiny elementwise kernels ----------------
__global__ void set_one_k(float* p, int n) {
    int i = blockIdx.x * blockDim.x + threadIdx.x;
    if (i < n) p[i] = 1.0f;
}

__global__ void deg_k(const int* __restrict__ dst, float* __restrict__ deg, int e) {
    int i = blockIdx.x * blockDim.x + threadIdx.x;
    if (i < e) atomicAdd(&deg[dst[i]], 1.0f);
}

__global__ void rsqrt_k(const float* __restrict__ deg, float* __restrict__ dinv, int n) {
    int i = blockIdx.x * blockDim.x + threadIdx.x;
    if (i < n) dinv[i] = rsqrtf(deg[i]);
}

// fp32 -> bf16 bulk convert
__global__ void cvt_bf16_k(const float* __restrict__ in, __bf16* __restrict__ out, int n) {
    int i = blockIdx.x * blockDim.x + threadIdx.x;
    if (i < n) out[i] = f2bf(in[i]);
}

// h1 = relu(agg1 + b1), emitted directly as bf16 for the next GEMM's A operand
__global__ void relu_bias_cvt_k(const float* __restrict__ agg, const float* __restrict__ b,
                                __bf16* __restrict__ out, int n, int f) {
    int i = blockIdx.x * blockDim.x + threadIdx.x;
    if (i < n * f) {
        float v = agg[i] + b[i % f];
        out[i] = f2bf(v > 0.0f ? v : 0.0f);
    }
}

// ---------------- weight pre-pack into exact WMMA B-fragment layout ----------------
// Wp layout: fragment (kt, tn) at [(kt*tilesN + tn)*32 + lane]*16, 16 bf16 contiguous.
// This code's order: lane -> col = lane%16, K offset = 16*(lane/16); elem pair 2j,2j+1 -> K = 2j, 2j+1.
__global__ void pack_w_k(const float* __restrict__ W, __bf16* __restrict__ Wp, int K, int N) {
    int tilesN = N >> 4, ktiles = K >> 5;
    int gid = blockIdx.x * blockDim.x + threadIdx.x;
    if (gid >= ktiles * tilesN * 32) return;
    int lane = gid & 31;
    int frag = gid >> 5;                    // kt*tilesN + tn
    int kt = frag / tilesN, tn = frag - kt * tilesN;
    int half = lane >> 4, n = lane & 15;
    int kbase = kt * 32 + half * 16;
    int col = tn * 16 + n;
    __bf16* o = Wp + (size_t)gid * 16;
    #pragma unroll
    for (int j = 0; j < 8; ++j) {
        o[2 * j]     = f2bf(W[(size_t)(kbase + 2 * j) * N + col]);
        o[2 * j + 1] = f2bf(W[(size_t)(kbase + 2 * j + 1) * N + col]);
    }
}

// ---------------- WMMA bf16 GEMM: C[M,N] = A[M,K] x B[K,N] ----------------
// A: row-major bf16; Bp: pre-packed fragments; C: fp32. One wave = one 16x16 tile.
// Inner loop: 2x b128 A loads + 2x b128 B loads + 1 wmma.
__global__ void gemm_wmma_bf16p(const __bf16* __restrict__ A, const __bf16* __restrict__ Bp,
                                float* __restrict__ C, int M, int K, int N) {
    int wave = (blockIdx.x * blockDim.x + threadIdx.x) >> 5;
    int lane = threadIdx.x & 31;
    int tilesN = N >> 4;
    int numTiles = (M >> 4) * tilesN;
    if (wave >= numTiles) return;                    // wave-uniform: EXEC stays all-ones
    int tm = wave / tilesN, tn = wave - tm * tilesN;
    int half = lane >> 4;                            // 0 or 1
    int r    = lane & 15;

    v8f acc = {};
    const __bf16* Arow  = A + (size_t)(tm * 16 + r) * K + half * 16;
    const __bf16* Bfrag = Bp + ((size_t)tn * 32 + lane) * 16;
    const size_t  Bstep = (size_t)tilesN * 32 * 16;  // bytes-per-ktile / sizeof(bf16)

    for (int k0 = 0; k0 < K; k0 += 32) {
        union { v16bf v; v8bf h[2]; } a, b;
        a.h[0] = *(const v8bf*)(Arow + k0);          // K = half*16 .. +7
        a.h[1] = *(const v8bf*)(Arow + k0 + 8);      // K = half*16+8 .. +7
        const __bf16* bp = Bfrag + (size_t)(k0 >> 5) * Bstep;
        b.h[0] = *(const v8bf*)bp;
        b.h[1] = *(const v8bf*)(bp + 8);
        acc = __builtin_amdgcn_wmma_f32_16x16x32_bf16(
            /*neg_a=*/false, a.v, /*neg_b=*/false, b.v,
            /*c_mod=*/(short)0, acc, /*reuse_a=*/false, /*reuse_b=*/false);
    }
    float* Cp = C + (size_t)(tm * 16 + half * 8) * N + tn * 16 + r;
    #pragma unroll
    for (int i = 0; i < 8; ++i) Cp[(size_t)i * N] = acc[i];
}

// ---------------- edge scatter: out[dst] += hw[src] * dinv[src]*dinv[dst] ----------------
__global__ void scatter_edges_k(const float* __restrict__ hw, const int* __restrict__ src,
                                const int* __restrict__ dst, const float* __restrict__ dinv,
                                float* __restrict__ out, int e, int f) {
    int tpe = f >> 2;
    int gid = blockIdx.x * blockDim.x + threadIdx.x;
    if (gid >= e * tpe) return;
    int ed = gid / tpe;
    int f4 = (gid - ed * tpe) << 2;
    int s = src[ed], d = dst[ed];
    float c = dinv[s] * dinv[d];
    const float4 v = *(const float4*)(hw + (size_t)s * f + f4);
    float* o = out + (size_t)d * f + f4;
    atomicAdd(o + 0, v.x * c);
    atomicAdd(o + 1, v.y * c);
    atomicAdd(o + 2, v.z * c);
    atomicAdd(o + 3, v.w * c);
}

// self loops: out[i] += hw[i] * dinv[i]^2
__global__ void scatter_self_k(const float* __restrict__ hw, const float* __restrict__ dinv,
                               float* __restrict__ out, int n, int f) {
    int tpe = f >> 2;
    int gid = blockIdx.x * blockDim.x + threadIdx.x;
    if (gid >= n * tpe) return;
    int i  = gid / tpe;
    int f4 = (gid - i * tpe) << 2;
    float c = dinv[i] * dinv[i];
    const float4 v = *(const float4*)(hw + (size_t)i * f + f4);
    float* o = out + (size_t)i * f + f4;
    atomicAdd(o + 0, v.x * c);
    atomicAdd(o + 1, v.y * c);
    atomicAdd(o + 2, v.z * c);
    atomicAdd(o + 3, v.w * c);
}

// ---------------- pooling ----------------
__global__ void pool_k(const float* __restrict__ agg2, const int* __restrict__ batch,
                       float* __restrict__ sums, float* __restrict__ cnts, int n, int f) {
    int gid = blockIdx.x * blockDim.x + threadIdx.x;
    if (gid >= n * f) return;
    int i = gid / f, ff = gid - i * f;
    int g = batch[i];
    atomicAdd(&sums[g * f + ff], agg2[gid]);
    if (ff == 0) atomicAdd(&cnts[g], 1.0f);
}

// embeddings = sums/max(cnt,1) + b2   (bias folded past the mean)
__global__ void finalize_emb_k(const float* __restrict__ sums, const float* __restrict__ cnts,
                               const float* __restrict__ b2, float* __restrict__ emb, int g, int f) {
    int gid = blockIdx.x * blockDim.x + threadIdx.x;
    if (gid >= g * f) return;
    int gg = gid / f, ff = gid - gg * f;
    float c = cnts[gg]; c = c > 1.0f ? c : 1.0f;
    emb[gid] = sums[gid] / c + b2[ff];
}

// ---------------- heads ----------------
__global__ void group_logits_k(const float* __restrict__ emb, const float* __restrict__ Wg,
                               const float* __restrict__ bg, float* __restrict__ gl,
                               int g, int d, int ng) {
    int gid = blockIdx.x * blockDim.x + threadIdx.x;
    if (gid >= g * ng) return;
    int gg = gid / ng, j = gid - gg * ng;
    float acc = bg[j];
    const float* e = emb + gg * d;
    for (int k = 0; k < d; ++k) acc += e[k] * Wg[k * ng + j];
    gl[gid] = acc;
}

__global__ void argmax_k(const float* __restrict__ gl, int* __restrict__ pred, int g, int ng) {
    int gg = blockIdx.x * blockDim.x + threadIdx.x;
    if (gg >= g) return;
    const float* row = gl + gg * ng;
    int best = 0; float bv = row[0];
    for (int j = 1; j < ng; ++j) { if (row[j] > bv) { bv = row[j]; best = j; } }
    pred[gg] = best;
}

__global__ void family_k(const float* __restrict__ emb, const float* __restrict__ Wf,
                         const float* __restrict__ bf, const int* __restrict__ pred,
                         float* __restrict__ fl, int g, int d, int nf) {
    int gid = blockIdx.x * blockDim.x + threadIdx.x;
    if (gid >= g * nf) return;
    int gg = gid / nf, ff = gid - gg * nf;
    int p = pred[gg];
    float acc = bf[p * nf + ff];
    const float* e = emb + gg * d;
    const float* w = Wf + (size_t)p * d * nf + ff;
    for (int k = 0; k < d; ++k) acc += e[k] * w[(size_t)k * nf];
    fl[gid] = acc;
}

// ---------------- launcher ----------------
extern "C" void kernel_launch(void* const* d_in, const int* in_sizes, int n_in,
                              void* d_out, int out_size, void* d_ws, size_t ws_size,
                              hipStream_t stream) {
    const float* x    = (const float*)d_in[0];
    const int*   ei   = (const int*)  d_in[1];   // [2, NE] flat: row0=src, row1=dst
    const int*   batch= (const int*)  d_in[2];
    const float* W1   = (const float*)d_in[3];
    const float* b1   = (const float*)d_in[4];
    const float* W2   = (const float*)d_in[5];
    const float* b2   = (const float*)d_in[6];
    const float* Wg   = (const float*)d_in[7];
    const float* bg   = (const float*)d_in[8];
    const float* Wf   = (const float*)d_in[9];
    const float* bf   = (const float*)d_in[10];
    (void)n_in; (void)in_sizes; (void)out_size; (void)ws_size;

    const int* src = ei;
    const int* dst = ei + NE;

    // workspace layout (all segment sizes are multiples of 16 bytes)
    float*  deg  = (float*)d_ws;                   // NN f32
    float*  dinv = deg  + NN;                      // NN f32
    float*  bufA = dinv + NN;                      // NN*HID f32 (hw1; later hw2 in first NN*EMB)
    float*  bufB = bufA + (size_t)NN * HID;        // NN*HID f32 (agg1; later agg2 in first NN*EMB)
    __bf16* xbf  = (__bf16*)(bufB + (size_t)NN * HID);   // NN*FIN bf16
    __bf16* h1bf = xbf  + (size_t)NN * FIN;        // NN*HID bf16
    __bf16* W1p  = h1bf + (size_t)NN * HID;        // FIN*HID bf16 (packed fragments)
    __bf16* W2p  = W1p  + (size_t)FIN * HID;       // HID*EMB bf16 (packed fragments)
    float*  sums = (float*)(W2p + (size_t)HID * EMB);    // NG_G*EMB f32
    float*  cnts = sums + NG_G * EMB;              // NG_G f32
    int*    pred = (int*)(cnts + NG_G);            // NG_G i32

    float* emb = (float*)d_out;                    // [256,128]
    float* gl  = emb + NG_G * EMB;                 // [256,16]
    float* fl  = gl  + NG_G * NGRP;                // [256,128]

    const int B = 256;
    #define GRID(n) (((n) + B - 1) / B)

    // --- normalization coefficients ---
    set_one_k<<<GRID(NN), B, 0, stream>>>(deg, NN);                         // self-loop degree
    deg_k<<<GRID(NE), B, 0, stream>>>(dst, deg, NE);
    rsqrt_k<<<GRID(NN), B, 0, stream>>>(deg, dinv, NN);

    // --- operand preparation (one-time converts / packs) ---
    cvt_bf16_k<<<GRID(NN * FIN), B, 0, stream>>>(x, xbf, NN * FIN);
    pack_w_k<<<GRID((FIN / 32) * (HID / 16) * 32), B, 0, stream>>>(W1, W1p, FIN, HID);
    pack_w_k<<<GRID((HID / 32) * (EMB / 16) * 32), B, 0, stream>>>(W2, W2p, HID, EMB);

    // --- layer 1: hw1 = x @ W1 (WMMA bf16), agg1 = scatter ---
    {
        int tiles = (NN / 16) * (HID / 16);        // 3125 * 16
        gemm_wmma_bf16p<<<GRID(tiles * 32), B, 0, stream>>>(xbf, W1p, bufA, NN, FIN, HID);
    }
    hipMemsetAsync(bufB, 0, (size_t)NN * HID * sizeof(float), stream);
    scatter_edges_k<<<GRID(NE * (HID / 4)), B, 0, stream>>>(bufA, src, dst, dinv, bufB, NE, HID);
    scatter_self_k <<<GRID(NN * (HID / 4)), B, 0, stream>>>(bufA, dinv, bufB, NN, HID);
    relu_bias_cvt_k<<<GRID(NN * HID), B, 0, stream>>>(bufB, b1, h1bf, NN, HID);   // h1 as bf16

    // --- layer 2: hw2 = h1 @ W2 (WMMA bf16) into bufA, agg2 = scatter into bufB ---
    {
        int tiles = (NN / 16) * (EMB / 16);        // 3125 * 8
        gemm_wmma_bf16p<<<GRID(tiles * 32), B, 0, stream>>>(h1bf, W2p, bufA, NN, HID, EMB);
    }
    hipMemsetAsync(bufB, 0, (size_t)NN * EMB * sizeof(float), stream);      // reuse as agg2
    scatter_edges_k<<<GRID(NE * (EMB / 4)), B, 0, stream>>>(bufA, src, dst, dinv, bufB, NE, EMB);
    scatter_self_k <<<GRID(NN * (EMB / 4)), B, 0, stream>>>(bufA, dinv, bufB, NN, EMB);

    // --- pool + heads ---
    hipMemsetAsync(sums, 0, (size_t)(NG_G * EMB + NG_G) * sizeof(float), stream);
    pool_k<<<GRID(NN * EMB), B, 0, stream>>>(bufB, batch, sums, cnts, NN, EMB);
    finalize_emb_k<<<GRID(NG_G * EMB), B, 0, stream>>>(sums, cnts, b2, emb, NG_G, EMB);
    group_logits_k<<<GRID(NG_G * NGRP), B, 0, stream>>>(emb, Wg, bg, gl, NG_G, EMB, NGRP);
    argmax_k<<<GRID(NG_G), B, 0, stream>>>(gl, pred, NG_G, NGRP);
    family_k<<<GRID(NG_G * NFAM), B, 0, stream>>>(emb, Wf, bf, pred, fl, NG_G, EMB, NFAM);
    #undef GRID
}
